// Net_GAT_45148696215618
// MI455X (gfx1250) — matrix operand order repaired and checked by reading, not compiled, not run on hardware
//
#include <hip/hip_runtime.h>
#include <hip/hip_bf16.h>

typedef __attribute__((ext_vector_type(16))) __bf16 v16bf;
typedef __attribute__((ext_vector_type(8)))  float  v8f;

#define HEADS1 8
#define OUT1   8
#define F1     64      // HEADS1*OUT1
#define FIN    512
#define C2     40
#define C2P    48      // padded to 3x16 col tiles
#define NEG_SLOPE 0.2f
#define ENC_NEG_INF 0x007FFFFFu

__device__ __forceinline__ unsigned fenc(float f) {
    unsigned u = __float_as_uint(f);
    return (u & 0x80000000u) ? ~u : (u | 0x80000000u);
}
__device__ __forceinline__ float fdec(unsigned u) {
    u = (u & 0x80000000u) ? (u & 0x7FFFFFFFu) : ~u;
    return __uint_as_float(u);
}
__device__ __forceinline__ float lrelu(float v) { return v > 0.0f ? v : NEG_SLOPE * v; }

// ---------------- weight prep: transpose + split-bf16 (hi/lo) ----------------
__global__ void prep_w_kernel(const float* __restrict__ W1, const float* __restrict__ W2,
                              __bf16* __restrict__ w1hi, __bf16* __restrict__ w1lo,
                              __bf16* __restrict__ w2hi, __bf16* __restrict__ w2lo) {
    int tid = blockIdx.x * blockDim.x + threadIdx.x;
    if (tid < F1 * FIN) {                       // W1 [512][64] -> W1T [64][512]
        int n = tid / FIN, k = tid % FIN;
        float w = W1[(size_t)k * F1 + n];
        __bf16 hi = (__bf16)w;
        w1hi[tid] = hi;
        w1lo[tid] = (__bf16)(w - (float)hi);
    }
    if (tid < C2P * F1) {                       // W2 [64][40] -> W2T [48][64], zero-pad
        int n = tid / F1, k = tid % F1;
        float w = (n < C2) ? W2[(size_t)k * C2 + n] : 0.0f;
        __bf16 hi = (__bf16)w;
        w2hi[tid] = hi;
        w2lo[tid] = (__bf16)(w - (float)hi);
    }
}

// ---------------- workspace init (must re-init every call) ----------------
__global__ void init_kernel(float* __restrict__ agg1, unsigned* __restrict__ m1,
                            float* __restrict__ den1, float* __restrict__ agg2,
                            unsigned* __restrict__ m2, float* __restrict__ den2, int n) {
    long tid = (long)blockIdx.x * blockDim.x + threadIdx.x;
    if (tid < (long)n * F1) agg1[tid] = 0.0f;
    if (tid < (long)n * C2) agg2[tid] = 0.0f;
    if (tid < (long)n * HEADS1) { m1[tid] = ENC_NEG_INF; den1[tid] = 0.0f; }
    if (tid < n) { m2[tid] = ENC_NEG_INF; den2[tid] = 0.0f; }
}

// ---------------- WMMA GEMM: Y[n,ncols] = X[n,K] @ B^T (B stored [NT*16][K] bf16 hi/lo) ----
struct AFrag { v16bf hi, lo; };

__device__ __forceinline__ AFrag make_afrag(const float* __restrict__ p) {
    // elements 0..15 <- p[0..15]  (contiguous k order)
    float v[16];
    float4 f0 = *(const float4*)(p);
    float4 f1 = *(const float4*)(p + 4);
    float4 f2 = *(const float4*)(p + 8);
    float4 f3 = *(const float4*)(p + 12);
    v[0]=f0.x; v[1]=f0.y; v[2]=f0.z; v[3]=f0.w;
    v[4]=f1.x; v[5]=f1.y; v[6]=f1.z; v[7]=f1.w;
    v[8]=f2.x; v[9]=f2.y; v[10]=f2.z; v[11]=f2.w;
    v[12]=f3.x; v[13]=f3.y; v[14]=f3.z; v[15]=f3.w;
    AFrag a;
#pragma unroll
    for (int i = 0; i < 16; ++i) {
        __bf16 h = (__bf16)v[i];
        a.hi[i] = h;
        a.lo[i] = (__bf16)(v[i] - (float)h);
    }
    return a;
}

template <int K, int NT>
__global__ void gemm_wmma_kernel(const float* __restrict__ X,
                                 const __bf16* __restrict__ Bhi,
                                 const __bf16* __restrict__ Blo,
                                 float* __restrict__ Y,
                                 int n, int ldy, int ncols) {
    int lane = threadIdx.x & 31;
    int wave = threadIdx.x >> 5;
    int r0 = (blockIdx.x * 8 + wave) * 16;
    if (r0 >= n) return;                       // wave-uniform exit: EXEC stays all-ones

    int arow = r0 + (lane & 15);
    if (arow >= n) arow = n - 1;               // clamp (not stored)
    const int kA_off = (lane & 16) ? 16 : 0;
    const int kB_off = (lane & 16) ? 16 : 0;
    const int ncol = lane & 15;

    v8f acc[NT];
#pragma unroll
    for (int t = 0; t < NT; ++t)
#pragma unroll
        for (int i = 0; i < 8; ++i) acc[t][i] = 0.0f;

    for (int k0 = 0; k0 < K; k0 += 32) {
        AFrag a = make_afrag(X + (size_t)arow * K + k0 + kA_off);
#pragma unroll
        for (int t = 0; t < NT; ++t) {
            size_t boff = (size_t)(t * 16 + ncol) * K + k0 + kB_off;
            v16bf bhi = *(const v16bf*)(Bhi + boff);
            v16bf blo = *(const v16bf*)(Blo + boff);
            acc[t] = __builtin_amdgcn_wmma_f32_16x16x32_bf16(false, a.hi, false, blo,
                                                             (short)0, acc[t], false, false);
            acc[t] = __builtin_amdgcn_wmma_f32_16x16x32_bf16(false, a.lo, false, bhi,
                                                             (short)0, acc[t], false, false);
            acc[t] = __builtin_amdgcn_wmma_f32_16x16x32_bf16(false, a.hi, false, bhi,
                                                             (short)0, acc[t], false, false);
        }
    }

    int rbase = r0 + ((lane & 16) ? 8 : 0);
#pragma unroll
    for (int t = 0; t < NT; ++t) {
        int col = t * 16 + ncol;
        if (col >= ncols) continue;
#pragma unroll
        for (int v = 0; v < 8; ++v) {
            int row = rbase + v;
            if (row < n) Y[(size_t)row * ldy + col] = acc[t][v];
        }
    }
}

// ---------------- per-node attention logits ----------------
__global__ void logits1_kernel(const float* __restrict__ h1, const float* __restrict__ asrc,
                               const float* __restrict__ adst, float* __restrict__ als,
                               float* __restrict__ ald, int n) {
    int tid = blockIdx.x * blockDim.x + threadIdx.x;
    if (tid >= n * HEADS1) return;
    int i = tid / HEADS1, h = tid % HEADS1;
    const float* hp = h1 + (size_t)i * F1 + h * OUT1;
    float4 a = *(const float4*)hp;
    float4 b = *(const float4*)(hp + 4);
    const float* as = asrc + h * OUT1;
    const float* ad = adst + h * OUT1;
    als[tid] = a.x*as[0] + a.y*as[1] + a.z*as[2] + a.w*as[3]
             + b.x*as[4] + b.y*as[5] + b.z*as[6] + b.w*as[7];
    ald[tid] = a.x*ad[0] + a.y*ad[1] + a.z*ad[2] + a.w*ad[3]
             + b.x*ad[4] + b.y*ad[5] + b.z*ad[6] + b.w*ad[7];
}

__global__ void logits2_kernel(const float* __restrict__ h2, const float* __restrict__ asrc,
                               const float* __restrict__ adst, float* __restrict__ als,
                               float* __restrict__ ald, int n) {
    int i = blockIdx.x * blockDim.x + threadIdx.x;
    if (i >= n) return;
    const float* hp = h2 + (size_t)i * C2;
    float s = 0.0f, d = 0.0f;
#pragma unroll
    for (int c = 0; c < C2; ++c) { float v = hp[c]; s += v * asrc[c]; d += v * adst[c]; }
    als[i] = s; ald[i] = d;
}

// ---------------- layer-1 edge passes (thread per (edge, head)) ----------------
__global__ void edge_max1_kernel(const int* __restrict__ ei, const float* __restrict__ als,
                                 const float* __restrict__ ald, unsigned* __restrict__ m1,
                                 int E, int n) {
    long tid = (long)blockIdx.x * blockDim.x + threadIdx.x;
    if (tid >= (long)(E + n) * HEADS1) return;
    int e = (int)(tid >> 3), h = (int)(tid & 7);
    int src, dst;
    if (e < E) { src = ei[e]; dst = ei[E + e]; } else { src = dst = e - E; }
    float v = lrelu(als[src * HEADS1 + h] + ald[dst * HEADS1 + h]);
    atomicMax(&m1[dst * HEADS1 + h], fenc(v));
}

__global__ void edge_agg1_kernel(const int* __restrict__ ei, const float* __restrict__ als,
                                 const float* __restrict__ ald, const unsigned* __restrict__ m1,
                                 float* __restrict__ den1, const float* __restrict__ h1,
                                 float* __restrict__ agg1, int E, int n) {
    long tid = (long)blockIdx.x * blockDim.x + threadIdx.x;
    if (tid >= (long)(E + n) * HEADS1) return;
    int e = (int)(tid >> 3), h = (int)(tid & 7);
    int src, dst;
    if (e < E) { src = ei[e]; dst = ei[E + e]; } else { src = dst = e - E; }
    float v = lrelu(als[src * HEADS1 + h] + ald[dst * HEADS1 + h]);
    float ex = expf(v - fdec(m1[dst * HEADS1 + h]));
    atomicAdd(&den1[dst * HEADS1 + h], ex);
    const float* hp = h1 + (size_t)src * F1 + h * OUT1;
    float4 p = *(const float4*)hp;
    float4 q = *(const float4*)(hp + 4);
    float* op = agg1 + (size_t)dst * F1 + h * OUT1;
    atomicAdd(op + 0, p.x * ex); atomicAdd(op + 1, p.y * ex);
    atomicAdd(op + 2, p.z * ex); atomicAdd(op + 3, p.w * ex);
    atomicAdd(op + 4, q.x * ex); atomicAdd(op + 5, q.y * ex);
    atomicAdd(op + 6, q.z * ex); atomicAdd(op + 7, q.w * ex);
}

__global__ void normalize1_kernel(float* __restrict__ agg1, const float* __restrict__ den1,
                                  const float* __restrict__ b1, int n) {
    long tid = (long)blockIdx.x * blockDim.x + threadIdx.x;
    if (tid >= (long)n * F1) return;
    int i = (int)(tid / F1), c = (int)(tid % F1);
    float v = agg1[tid] / den1[i * HEADS1 + (c >> 3)] + b1[c];
    agg1[tid] = lrelu(v);                      // inter-layer activation, in place
}

// ---------------- layer-2 edge passes ----------------
__global__ void edge_max2_kernel(const int* __restrict__ ei, const float* __restrict__ als,
                                 const float* __restrict__ ald, unsigned* __restrict__ m2,
                                 int E, int n) {
    long tid = (long)blockIdx.x * blockDim.x + threadIdx.x;
    if (tid >= (long)(E + n)) return;
    int e = (int)tid;
    int src, dst;
    if (e < E) { src = ei[e]; dst = ei[E + e]; } else { src = dst = e - E; }
    float v = lrelu(als[src] + ald[dst]);
    atomicMax(&m2[dst], fenc(v));
}

__global__ void edge_agg2_kernel(const int* __restrict__ ei, const float* __restrict__ als,
                                 const float* __restrict__ ald, const unsigned* __restrict__ m2,
                                 float* __restrict__ den2, const float* __restrict__ h2,
                                 float* __restrict__ agg2, int E, int n) {
    long tid = (long)blockIdx.x * blockDim.x + threadIdx.x;
    if (tid >= (long)(E + n) * 8) return;
    int e = (int)(tid >> 3), j = (int)(tid & 7);   // 8 threads/edge, 5 cols each
    int src, dst;
    if (e < E) { src = ei[e]; dst = ei[E + e]; } else { src = dst = e - E; }
    float v = lrelu(als[src] + ald[dst]);
    float ex = expf(v - fdec(m2[dst]));
    if (j == 0) atomicAdd(&den2[dst], ex);
    const float* hp = h2 + (size_t)src * C2 + j * 5;
    float* op = agg2 + (size_t)dst * C2 + j * 5;
#pragma unroll
    for (int c = 0; c < 5; ++c) atomicAdd(op + c, hp[c] * ex);
}

// ---------------- normalize layer-2 + log_softmax (wave per node) ----------------
__global__ void logsoftmax_kernel(const float* __restrict__ agg2, const float* __restrict__ den2,
                                  const float* __restrict__ b2, float* __restrict__ out, int n) {
    int lane = threadIdx.x & 31;
    int node = blockIdx.x * 8 + (threadIdx.x >> 5);
    if (node >= n) return;
    float dinv = 1.0f / den2[node];
    const float* gp = agg2 + (size_t)node * C2;
    float v0 = gp[lane] * dinv + b2[lane];                                      // cols 0..31
    float v1 = (lane < 8) ? (gp[32 + lane] * dinv + b2[32 + lane]) : -__builtin_inff();
    float mx = fmaxf(v0, v1);
#pragma unroll
    for (int o = 16; o >= 1; o >>= 1) mx = fmaxf(mx, __shfl_xor(mx, o, 32));
    float s = expf(v0 - mx) + ((lane < 8) ? expf(v1 - mx) : 0.0f);
#pragma unroll
    for (int o = 16; o >= 1; o >>= 1) s += __shfl_xor(s, o, 32);
    float lse = mx + logf(s);
    out[(size_t)node * C2 + lane] = v0 - lse;
    if (lane < 8) out[(size_t)node * C2 + 32 + lane] = v1 - lse;
}

// ---------------- launch ----------------
extern "C" void kernel_launch(void* const* d_in, const int* in_sizes, int n_in,
                              void* d_out, int out_size, void* d_ws, size_t ws_size,
                              hipStream_t stream) {
    const float* x   = (const float*)d_in[0];
    const int*   ei  = (const int*)d_in[1];
    const float* W1  = (const float*)d_in[2];
    const float* as1 = (const float*)d_in[3];
    const float* ad1 = (const float*)d_in[4];
    const float* b1  = (const float*)d_in[5];
    const float* W2  = (const float*)d_in[6];
    const float* as2 = (const float*)d_in[7];
    const float* ad2 = (const float*)d_in[8];
    const float* b2  = (const float*)d_in[9];

    const int n  = in_sizes[0] / FIN;    // 100000
    const int E  = in_sizes[1] / 2;      // 1600000
    const long ET = (long)E + n;

    char* ws = (char*)d_ws;
    size_t off = 0;
    auto alloc = [&](size_t bytes) -> void* {
        void* p = ws + off;
        off += (bytes + 255) & ~(size_t)255;
        return p;
    };
    __bf16* w1hi = (__bf16*)alloc((size_t)F1 * FIN * 2);
    __bf16* w1lo = (__bf16*)alloc((size_t)F1 * FIN * 2);
    __bf16* w2hi = (__bf16*)alloc((size_t)C2P * F1 * 2);
    __bf16* w2lo = (__bf16*)alloc((size_t)C2P * F1 * 2);
    float*  h1   = (float*)alloc((size_t)n * F1 * 4);
    float*  als1p= (float*)alloc((size_t)n * HEADS1 * 4);
    float*  ald1p= (float*)alloc((size_t)n * HEADS1 * 4);
    unsigned* m1 = (unsigned*)alloc((size_t)n * HEADS1 * 4);
    float*  den1 = (float*)alloc((size_t)n * HEADS1 * 4);
    float*  agg1 = (float*)alloc((size_t)n * F1 * 4);
    float*  h2   = (float*)alloc((size_t)n * C2 * 4);
    float*  als2p= (float*)alloc((size_t)n * 4);
    float*  ald2p= (float*)alloc((size_t)n * 4);
    unsigned* m2 = (unsigned*)alloc((size_t)n * 4);
    float*  den2 = (float*)alloc((size_t)n * 4);
    float*  agg2 = (float*)alloc((size_t)n * C2 * 4);

    const int BT = 256;
    prep_w_kernel<<<(F1 * FIN + BT - 1) / BT, BT, 0, stream>>>(W1, W2, w1hi, w1lo, w2hi, w2lo);
    init_kernel<<<(unsigned)(((long)n * F1 + BT - 1) / BT), BT, 0, stream>>>(
        agg1, m1, den1, agg2, m2, den2, n);

    // ---- layer 1 ----
    gemm_wmma_kernel<FIN, 4><<<(n + 127) / 128, BT, 0, stream>>>(x, w1hi, w1lo, h1, n, F1, F1);
    logits1_kernel<<<(n * HEADS1 + BT - 1) / BT, BT, 0, stream>>>(h1, as1, ad1, als1p, ald1p, n);
    edge_max1_kernel<<<(unsigned)((ET * HEADS1 + BT - 1) / BT), BT, 0, stream>>>(
        ei, als1p, ald1p, m1, E, n);
    edge_agg1_kernel<<<(unsigned)((ET * HEADS1 + BT - 1) / BT), BT, 0, stream>>>(
        ei, als1p, ald1p, m1, den1, h1, agg1, E, n);
    normalize1_kernel<<<(unsigned)(((long)n * F1 + BT - 1) / BT), BT, 0, stream>>>(
        agg1, den1, b1, n);

    // ---- layer 2 ----
    gemm_wmma_kernel<F1, 3><<<(n + 127) / 128, BT, 0, stream>>>(agg1, w2hi, w2lo, h2, n, C2, C2);
    logits2_kernel<<<(n + BT - 1) / BT, BT, 0, stream>>>(h2, as2, ad2, als2p, ald2p, n);
    edge_max2_kernel<<<(unsigned)((ET + BT - 1) / BT), BT, 0, stream>>>(ei, als2p, ald2p, m2, E, n);
    edge_agg2_kernel<<<(unsigned)((ET * 8 + BT - 1) / BT), BT, 0, stream>>>(
        ei, als2p, ald2p, m2, den2, h2, agg2, E, n);
    logsoftmax_kernel<<<(n + 7) / 8, BT, 0, stream>>>(agg2, den2, b2, (float*)d_out, n);
}
